// MultiHeadedAttentionWithAdapterCache_48722109005951
// MI455X (gfx1250) — matrix operand order repaired and checked by reading, not compiled, not run on hardware
//
#include <hip/hip_runtime.h>
#include <hip/hip_bf16.h>

typedef __attribute__((ext_vector_type(16))) _Float16 v16h;
typedef __attribute__((ext_vector_type(8)))  float    v8f;

// Problem constants
#define BB    8
#define SS    1024
#define DD    1024
#define HH    16
#define HD    64
#define LMAX  2048

// ---------------------------------------------------------------------------
// Transpose + convert: W (f32, k-major KxN) -> Wt (f16, n-major NxK).
// 64x64 tiles through LDS; coalesced read and write.
// ---------------------------------------------------------------------------
__global__ __launch_bounds__(256)
void wt_f16_kernel(const float* __restrict__ W, _Float16* __restrict__ Wt)
{
    __shared__ _Float16 tile[64][65];
    const int tid = threadIdx.x;
    const int nb = blockIdx.x * 64;
    const int kb = blockIdx.y * 64;

    #pragma unroll
    for (int p = 0; p < 16; ++p) {
        const int idx = p * 256 + tid;
        const int k = idx >> 6, n = idx & 63;
        tile[n][k] = (_Float16)W[(size_t)(kb + k) * DD + nb + n];
    }
    __syncthreads();
    #pragma unroll
    for (int p = 0; p < 16; ++p) {
        const int idx = p * 256 + tid;
        const int n = idx >> 6, k = idx & 63;
        Wt[(size_t)(nb + n) * DD + kb + k] = tile[n][k];
    }
}

// ---------------------------------------------------------------------------
// Elementwise convert f32 -> f16 (float4 vectorized; count must be /4)
// ---------------------------------------------------------------------------
__global__ __launch_bounds__(256)
void f32_to_f16_kernel(const float* __restrict__ x, _Float16* __restrict__ y)
{
    const int i = blockIdx.x * 256 + threadIdx.x;
    const float4 v = ((const float4*)x)[i];
    y[4 * i + 0] = (_Float16)v.x;
    y[4 * i + 1] = (_Float16)v.y;
    y[4 * i + 2] = (_Float16)v.z;
    y[4 * i + 3] = (_Float16)v.w;
}

// ---------------------------------------------------------------------------
// GEMM: out(f16) = Xh(f16, MxK) @ Wt^T(f16, NxK stored n-major) + bias
// M=8192, N=K=1024. One wave -> 32x64 tile (2 A-frags x 4 B-frags = 8 WMMA
// per k-step); block = 4 waves -> 128x64 tile.
// transpose_out == 0 : out laid out (B,H,S,HD)   [Q, K]
// transpose_out == 1 : out laid out (B,H,HD,S)   [V, transposed]
// ---------------------------------------------------------------------------
__global__ __launch_bounds__(128)
void proj_gemm_f16(const _Float16* __restrict__ Xh, const _Float16* __restrict__ Wt,
                   const float* __restrict__ bias, _Float16* __restrict__ out,
                   int transpose_out)
{
    const int lane = threadIdx.x & 31;
    const int wid  = threadIdx.x >> 5;
    const int half = lane >> 4;
    const int l16  = lane & 15;

    const int mbase = blockIdx.x * 128 + wid * 32;
    const int nbase = blockIdx.y * 64;

    const _Float16* arow0 = Xh + (size_t)(mbase + l16) * DD;
    const _Float16* arow1 = Xh + (size_t)(mbase + 16 + l16) * DD;

    v8f acc[2][4];
    #pragma unroll
    for (int r = 0; r < 2; ++r)
        #pragma unroll
        for (int ns = 0; ns < 4; ++ns)
            #pragma unroll
            for (int i = 0; i < 8; ++i) acc[r][ns][i] = 0.0f;

    for (int k0 = 0; k0 < DD; k0 += 32) {
        v16h a0, a1;
        {
            const _Float16* ap0 = arow0 + k0 + half * 8;
            const _Float16* ap1 = arow1 + k0 + half * 8;
            #pragma unroll
            for (int q = 0; q < 8; ++q) {
                a0[q]     = ap0[q];
                a0[8 + q] = ap0[16 + q];
                a1[q]     = ap1[q];
                a1[8 + q] = ap1[16 + q];
            }
        }
        #pragma unroll
        for (int ns = 0; ns < 4; ++ns) {
            const int n = nbase + ns * 16 + l16;
            const _Float16* bp = Wt + (size_t)n * DD + k0 + half * 8;
            v16h bf;
            #pragma unroll
            for (int e = 0; e < 8; ++e) { bf[e] = bp[e]; bf[8 + e] = bp[16 + e]; }
            acc[0][ns] = __builtin_amdgcn_wmma_f32_16x16x32_f16(
                false, a0, false, bf, (short)0, acc[0][ns], false, false);
            acc[1][ns] = __builtin_amdgcn_wmma_f32_16x16x32_f16(
                false, a1, false, bf, (short)0, acc[1][ns], false, false);
        }
    }

    // Store with bias; C/D layout: row m = i + 8*half, col n = l16
    #pragma unroll
    for (int r = 0; r < 2; ++r) {
        #pragma unroll
        for (int ns = 0; ns < 4; ++ns) {
            const int n  = nbase + ns * 16 + l16;
            const int h  = n >> 6;
            const int hd = n & 63;
            const float bval = bias[n];
            #pragma unroll
            for (int i = 0; i < 8; ++i) {
                const int mrow = mbase + r * 16 + half * 8 + i;
                const int b = mrow >> 10;                  // S = 1024
                const int s = mrow & 1023;
                const _Float16 v = (_Float16)(acc[r][ns][i] + bval);
                if (transpose_out == 0) {
                    out[(((size_t)(b * HH + h) * SS) + s) * HD + hd] = v;
                } else {
                    out[(((size_t)(b * HH + h) * HD) + hd) * SS + s] = v;
                }
            }
        }
    }
}

// ---------------------------------------------------------------------------
// Flash-style attention. Grid = B*H*(S/64) blocks, 4 waves/block,
// each wave owns 16 query rows and streams kv in chunks of 32.
// ---------------------------------------------------------------------------
__global__ __launch_bounds__(128)
void attn_flash(const _Float16* __restrict__ Qf, const _Float16* __restrict__ Kf,
                const _Float16* __restrict__ Vt, const int* __restrict__ mask,
                const float* __restrict__ syn, const float* __restrict__ alpha_p,
                _Float16* __restrict__ Of)
{
    __shared__ _Float16 plds[4][16 * 34];      // padded rows: bank-conflict free

    const int lane = threadIdx.x & 31;
    const int wid  = threadIdx.x >> 5;
    const int half = lane >> 4;
    const int l16  = lane & 15;

    const int blk = blockIdx.x;                // bh * 16 + qt
    const int qt  = blk & 15;
    const int bh  = blk >> 4;
    const int h   = bh & 15;
    const int b   = bh >> 4;

    const int qbase = qt * 64 + wid * 16;

    const float alpha = 1.0f / (1.0f + expf(-alpha_p[0]));
    const float sA = alpha * 0.125f;           // alpha / sqrt(HD=64)
    const float sB = 1.0f - alpha;
    const float L2E = 1.44269504088896340736f;

    // Q A-fragments for contraction k = 0..31 (qa0) and 32..63 (qa1)
    const _Float16* qp = Qf + ((size_t)bh * SS + (qbase + l16)) * HD;
    v16h qa0, qa1;
    #pragma unroll
    for (int q = 0; q < 8; ++q) {
        qa0[q]     = qp[half * 8 + q];
        qa0[8 + q] = qp[16 + half * 8 + q];
        qa1[q]     = qp[32 + half * 8 + q];
        qa1[8 + q] = qp[48 + half * 8 + q];
    }

    float mrun[8], lrun[8];
    v8f oacc[4];
    #pragma unroll
    for (int i = 0; i < 8; ++i) { mrun[i] = -3.0e38f; lrun[i] = 0.0f; }
    #pragma unroll
    for (int ns = 0; ns < 4; ++ns)
        #pragma unroll
        for (int i = 0; i < 8; ++i) oacc[ns][i] = 0.0f;

    const size_t kbase   = (size_t)bh * SS * HD;
    const size_t vbase   = (size_t)bh * HD * SS;
    const size_t synbase = (size_t)h * LMAX * LMAX;
    const size_t mbbase  = (size_t)b * SS * SS;

    for (int kv = 0; kv < SS; kv += 32) {
        // ---- score tiles: s[0] cols [kv, kv+16), s[1] cols [kv+16, kv+32)
        v8f s[2];
        #pragma unroll
        for (int t = 0; t < 2; ++t) {
            const int col = kv + t * 16 + l16;
            const _Float16* kp = Kf + kbase + (size_t)col * HD + half * 8;
            v16h bk0, bk1;
            #pragma unroll
            for (int e = 0; e < 8; ++e) { bk0[e] = kp[e]; bk0[8 + e] = kp[16 + e]; bk1[e] = kp[32 + e]; bk1[8 + e] = kp[48 + e]; }
            v8f c;
            #pragma unroll
            for (int i = 0; i < 8; ++i) c[i] = 0.0f;
            c = __builtin_amdgcn_wmma_f32_16x16x32_f16(false, qa0, false, bk0,
                                                       (short)0, c, false, false);
            c = __builtin_amdgcn_wmma_f32_16x16x32_f16(false, qa1, false, bk1,
                                                       (short)0, c, false, false);
            s[t] = c;
        }

        // ---- merge: alpha*scores/sqrt(hd) + (1-alpha)*syn, then mask
        float mer[2][8];
        #pragma unroll
        for (int t = 0; t < 2; ++t) {
            const int col = kv + t * 16 + l16;
            #pragma unroll
            for (int i = 0; i < 8; ++i) {
                const int row = qbase + half * 8 + i;
                const float sy = syn[synbase + (size_t)row * LMAX + col];
                const int   mk = mask[mbbase + (size_t)row * SS + col];
                const float v  = s[t][i] * sA + sB * sy;
                mer[t][i] = (mk == 0) ? -1.0e9f : v;
            }
        }

        // ---- online softmax (row stats across the 16 lanes of each half)
        float p[2][8];
        #pragma unroll
        for (int i = 0; i < 8; ++i) {
            float mx = fmaxf(mer[0][i], mer[1][i]);
            #pragma unroll
            for (int off = 8; off >= 1; off >>= 1)
                mx = fmaxf(mx, __shfl_xor(mx, off, 32));
            const float mnew  = fmaxf(mrun[i], mx);
            const float scale = exp2f((mrun[i] - mnew) * L2E);
            lrun[i] *= scale;
            #pragma unroll
            for (int ns = 0; ns < 4; ++ns) oacc[ns][i] *= scale;
            const float p0 = exp2f((mer[0][i] - mnew) * L2E);
            const float p1 = exp2f((mer[1][i] - mnew) * L2E);
            p[0][i] = p0; p[1][i] = p1;
            float rs = p0 + p1;
            #pragma unroll
            for (int off = 8; off >= 1; off >>= 1)
                rs += __shfl_xor(rs, off, 32);
            lrun[i] += rs;
            mrun[i]  = mnew;
        }

        // ---- C-layout -> A-layout for P via LDS (same-wave, DS is in-order)
        _Float16* pl = plds[wid];
        #pragma unroll
        for (int i = 0; i < 8; ++i) {
            pl[(half * 8 + i) * 34 + l16]      = (_Float16)p[0][i];
            pl[(half * 8 + i) * 34 + 16 + l16] = (_Float16)p[1][i];
        }
        v16h pa;
        const _Float16* pr = pl + l16 * 34;
        #pragma unroll
        for (int q = 0; q < 8; ++q) {
            pa[q]     = pr[half * 8 + q];
            pa[8 + q] = pr[16 + half * 8 + q];
        }

        // ---- O += P @ V  (V stored transposed: contiguous B fragments)
        #pragma unroll
        for (int ns = 0; ns < 4; ++ns) {
            const int hd = ns * 16 + l16;
            const _Float16* vp = Vt + vbase + (size_t)hd * SS + kv + half * 8;
            v16h bv;
            #pragma unroll
            for (int e = 0; e < 8; ++e) { bv[e] = vp[e]; bv[8 + e] = vp[16 + e]; }
            oacc[ns] = __builtin_amdgcn_wmma_f32_16x16x32_f16(
                false, pa, false, bv, (short)0, oacc[ns], false, false);
        }
    }

    // ---- normalize and store O as (B, S, D) f16 with feature n = h*64+hd
    #pragma unroll
    for (int ns = 0; ns < 4; ++ns) {
        const int n = h * HD + ns * 16 + l16;
        #pragma unroll
        for (int i = 0; i < 8; ++i) {
            const int srow = qbase + half * 8 + i;
            const float v  = oacc[ns][i] / lrun[i];
            Of[((size_t)b * SS + srow) * DD + n] = (_Float16)v;
        }
    }
}

// ---------------------------------------------------------------------------
// Output projection: out(f32) = O(f16, 8192x1024) @ Wot^T(f16 n-major) + bo
// Same 32x64-per-wave tiling as proj_gemm_f16.
// ---------------------------------------------------------------------------
__global__ __launch_bounds__(128)
void out_proj_gemm(const _Float16* __restrict__ A, const _Float16* __restrict__ Wt,
                   const float* __restrict__ bias, float* __restrict__ out)
{
    const int lane = threadIdx.x & 31;
    const int wid  = threadIdx.x >> 5;
    const int half = lane >> 4;
    const int l16  = lane & 15;

    const int mbase = blockIdx.x * 128 + wid * 32;
    const int nbase = blockIdx.y * 64;

    const _Float16* arow0 = A + (size_t)(mbase + l16) * DD;
    const _Float16* arow1 = A + (size_t)(mbase + 16 + l16) * DD;

    v8f acc[2][4];
    #pragma unroll
    for (int r = 0; r < 2; ++r)
        #pragma unroll
        for (int ns = 0; ns < 4; ++ns)
            #pragma unroll
            for (int i = 0; i < 8; ++i) acc[r][ns][i] = 0.0f;

    for (int k0 = 0; k0 < DD; k0 += 32) {
        v16h a0, a1;
        {
            const _Float16* ap0 = arow0 + k0 + half * 8;
            const _Float16* ap1 = arow1 + k0 + half * 8;
            #pragma unroll
            for (int q = 0; q < 8; ++q) {
                a0[q]     = ap0[q];
                a0[8 + q] = ap0[16 + q];
                a1[q]     = ap1[q];
                a1[8 + q] = ap1[16 + q];
            }
        }
        #pragma unroll
        for (int ns = 0; ns < 4; ++ns) {
            const int n = nbase + ns * 16 + l16;
            const _Float16* bp = Wt + (size_t)n * DD + k0 + half * 8;
            v16h bf;
            #pragma unroll
            for (int e = 0; e < 8; ++e) { bf[e] = bp[e]; bf[8 + e] = bp[16 + e]; }
            acc[0][ns] = __builtin_amdgcn_wmma_f32_16x16x32_f16(
                false, a0, false, bf, (short)0, acc[0][ns], false, false);
            acc[1][ns] = __builtin_amdgcn_wmma_f32_16x16x32_f16(
                false, a1, false, bf, (short)0, acc[1][ns], false, false);
        }
    }

    #pragma unroll
    for (int r = 0; r < 2; ++r) {
        #pragma unroll
        for (int ns = 0; ns < 4; ++ns) {
            const int n = nbase + ns * 16 + l16;
            const float bval = bias[n];
            #pragma unroll
            for (int i = 0; i < 8; ++i) {
                const int mrow = mbase + r * 16 + half * 8 + i;
                out[(size_t)mrow * DD + n] = acc[r][ns][i] + bval;
            }
        }
    }
}

// ---------------------------------------------------------------------------
extern "C" void kernel_launch(void* const* d_in, const int* in_sizes, int n_in,
                              void* d_out, int out_size, void* d_ws, size_t ws_size,
                              hipStream_t stream)
{
    const float* query  = (const float*)d_in[0];
    const float* key_   = (const float*)d_in[1];
    const float* value  = (const float*)d_in[2];
    const int*   mask   = (const int*)  d_in[3];
    const float* Wq     = (const float*)d_in[4];
    const float* bq     = (const float*)d_in[5];
    const float* Wk     = (const float*)d_in[6];
    const float* bk     = (const float*)d_in[7];
    const float* Wv     = (const float*)d_in[8];
    const float* bv     = (const float*)d_in[9];
    const float* Wo     = (const float*)d_in[10];
    const float* bo     = (const float*)d_in[11];
    const float* syn    = (const float*)d_in[12];
    const float* alphap = (const float*)d_in[13];
    float* out = (float*)d_out;

    char* ws = (char*)d_ws;
    const size_t SEG = (size_t)16 * 1024 * 1024;     // 16 MB per f16 tensor
    _Float16* Qf  = (_Float16*)(ws + 0 * SEG);       // (B,H,S,HD)
    _Float16* Kf  = (_Float16*)(ws + 1 * SEG);       // (B,H,S,HD)
    _Float16* Vt  = (_Float16*)(ws + 2 * SEG);       // (B,H,HD,S)
    _Float16* Of  = (_Float16*)(ws + 3 * SEG);       // (B,S,D)
    _Float16* Xh  = (_Float16*)(ws + 4 * SEG);       // reused activation f16
    _Float16* Wtb = (_Float16*)(ws + 5 * SEG);       // reused weight f16 (2 MB)

    const dim3 gW(16, 16), bW(256);                  // weight transpose
    const dim3 gC(BB * SS * DD / (4 * 256)), bC(256);// f32->f16 convert
    const dim3 gG(64, 16), bG(128);                  // GEMMs (128x64 tiles)

    // Q projection
    wt_f16_kernel<<<gW, bW, 0, stream>>>(Wq, Wtb);
    f32_to_f16_kernel<<<gC, bC, 0, stream>>>(query, Xh);
    proj_gemm_f16<<<gG, bG, 0, stream>>>(Xh, Wtb, bq, Qf, 0);
    // K projection
    wt_f16_kernel<<<gW, bW, 0, stream>>>(Wk, Wtb);
    f32_to_f16_kernel<<<gC, bC, 0, stream>>>(key_, Xh);
    proj_gemm_f16<<<gG, bG, 0, stream>>>(Xh, Wtb, bk, Kf, 0);
    // V projection (transposed output)
    wt_f16_kernel<<<gW, bW, 0, stream>>>(Wv, Wtb);
    f32_to_f16_kernel<<<gC, bC, 0, stream>>>(value, Xh);
    proj_gemm_f16<<<gG, bG, 0, stream>>>(Xh, Wtb, bv, Vt, 1);

    attn_flash<<<BB * HH * (SS / 64), bG, 0, stream>>>(Qf, Kf, Vt, mask, syn,
                                                       alphap, Of);

    wt_f16_kernel<<<gW, bW, 0, stream>>>(Wo, Wtb);
    out_proj_gemm<<<gG, bG, 0, stream>>>(Of, Wtb, bo, out);
}
